// PatchedBertSelfAttention_24446953849301
// MI455X (gfx1250) — matrix operand order repaired and checked by reading, not compiled, not run on hardware
//
#include <hip/hip_runtime.h>
#include <hip/hip_bf16.h>

#define SLEN 2048
#define NB   2
#define NH   16
#define DH   64
#define HIDN 1024
#define NPOS 4095   // 2*MAX_POS-1 rows in dist_emb

typedef __attribute__((ext_vector_type(16))) _Float16 v16h;
typedef __attribute__((ext_vector_type(8)))  float    v8f;

union U32H2 { unsigned u; _Float16 h[2]; };

__device__ __forceinline__ v8f wmma16(v16h a, v16h b, v8f c) {
  // v_wmma_f32_16x16x32_f16 : D = A(16x32) x B(32x16) + C
  return __builtin_amdgcn_wmma_f32_16x16x32_f16(false, a, false, b, (short)0, c,
                                                false, false);
}

// ---- fragment loaders (wave32; contiguous k order, same for A and B) ----------------
// A 16x32 f16: lane m = lane&15; lane-group (lane>>4) selects K in [0,16)/[16,32).
// VGPR p (pair 2p,2p+1): K = group*16 + p*2.
__device__ __forceinline__ v16h load_a_frag(const _Float16* src, int stride, int lane) {
  int m = lane & 15, hi = lane >> 4;
  v16h a;
#pragma unroll
  for (int p = 0; p < 8; ++p) {
    int k0 = hi * 16 + p * 2;
    U32H2 u; u.u = *(const unsigned*)(src + (size_t)m * stride + k0);
    a[2 * p] = u.h[0]; a[2 * p + 1] = u.h[1];
  }
  return a;
}

// B 32x16 f16 from source stored [n][k] row-major (B[k][n] = src[n][k]).
__device__ __forceinline__ v16h load_b_nk(const _Float16* src, int stride, int lane) {
  int n = lane & 15, khi = (lane >> 4) * 16;
  v16h bf;
#pragma unroll
  for (int p = 0; p < 8; ++p) {
    U32H2 u; u.u = *(const unsigned*)(src + (size_t)n * stride + khi + 2 * p);
    bf[2 * p] = u.h[0]; bf[2 * p + 1] = u.h[1];
  }
  return bf;
}

// B 32x16 f16 from source stored [k][n] row-major (V / V2 tiles).
__device__ __forceinline__ v16h load_b_kn(const _Float16* src, int stride, int lane) {
  int n = lane & 15, khi = (lane >> 4) * 16;
  v16h bf;
#pragma unroll
  for (int p = 0; p < 8; ++p) {
    bf[2 * p]     = src[(size_t)(khi + 2 * p)     * stride + n];
    bf[2 * p + 1] = src[(size_t)(khi + 2 * p + 1) * stride + n];
  }
  return bf;
}

// B fragment from f16 dist_emb rows [baserow+c], c = column, with row clamp.
__device__ __forceinline__ v16h load_b_erows(const _Float16* E, int baserow, int koff,
                                             int lane) {
  int c = lane & 15, khi = (lane >> 4) * 16;
  int row = baserow + c;
  row = row < 0 ? 0 : (row > (NPOS - 1) ? (NPOS - 1) : row);
  const _Float16* src = E + (size_t)row * DH + koff;
  v16h bf;
#pragma unroll
  for (int p = 0; p < 8; ++p) {
    U32H2 u; u.u = *(const unsigned*)(src + khi + 2 * p);
    bf[2 * p] = u.h[0]; bf[2 * p + 1] = u.h[1];
  }
  return bf;
}

// ---- kernel 0: dist_emb f32 -> f16 ----------------------------------------
__global__ void cvt_e_kernel(const float* __restrict__ e, _Float16* __restrict__ o,
                             int n) {
  int i = blockIdx.x * blockDim.x + threadIdx.x;
  if (i < n) o[i] = (_Float16)e[i];
}

// ---- kernel 1: fused QKVV2 projection (X @ W^T + b) -> f16 [B,H,S,D] -------
__global__ __launch_bounds__(128) void proj_kernel(
    const float* __restrict__ X,
    const float* __restrict__ W0, const float* __restrict__ bb0,
    const float* __restrict__ W1, const float* __restrict__ bb1,
    const float* __restrict__ W2, const float* __restrict__ bb2,
    const float* __restrict__ W3, const float* __restrict__ bb3,
    _Float16* __restrict__ O0, _Float16* __restrict__ O1,
    _Float16* __restrict__ O2, _Float16* __restrict__ O3) {
  const int z = blockIdx.z;
  const float* W    = (z == 0) ? W0 : (z == 1) ? W1 : (z == 2) ? W2 : W3;
  const float* bias = (z == 0) ? bb0 : (z == 1) ? bb1 : (z == 2) ? bb2 : bb3;
  _Float16* O       = (z == 0) ? O0 : (z == 1) ? O1 : (z == 2) ? O2 : O3;

  __shared__ _Float16 Xt[64 * 32];
  __shared__ _Float16 Wt[64 * 32];

  const int tid = threadIdx.x;
  const int lane = tid & 31, wv = tid >> 5;
  const int m0 = blockIdx.x * 64;   // rows of X (b*S+s)
  const int n0 = blockIdx.y * 64;   // output channel

  v8f acc0 = {}, acc1 = {}, acc2 = {}, acc3 = {};

  for (int k0 = 0; k0 < HIDN; k0 += 32) {
    __syncthreads();
#pragma unroll
    for (int i = 0; i < 16; ++i) {  // 2048 elements / 128 threads
      int idx = tid + i * 128;
      int rl = idx >> 5, kl = idx & 31;
      Xt[idx] = (_Float16)X[(size_t)(m0 + rl) * HIDN + k0 + kl];
      Wt[idx] = (_Float16)W[(size_t)(n0 + rl) * HIDN + k0 + kl];
    }
    __syncthreads();
    v16h a = load_a_frag(Xt + wv * 16 * 32, 32, lane);
    acc0 = wmma16(a, load_b_nk(Wt + 0 * 16 * 32, 32, lane), acc0);
    acc1 = wmma16(a, load_b_nk(Wt + 1 * 16 * 32, 32, lane), acc1);
    acc2 = wmma16(a, load_b_nk(Wt + 2 * 16 * 32, 32, lane), acc2);
    acc3 = wmma16(a, load_b_nk(Wt + 3 * 16 * 32, 32, lane), acc3);
  }

  v8f accs[4] = {acc0, acc1, acc2, acc3};
#pragma unroll
  for (int sub = 0; sub < 4; ++sub) {
    int n = n0 + sub * 16 + (lane & 15);
    float bvv = bias[n];
    int h = n >> 6, dd = n & 63;
#pragma unroll
    for (int r = 0; r < 8; ++r) {
      int m = m0 + wv * 16 + (lane >> 4) * 8 + r;
      int bb = m >> 11, s = m & (SLEN - 1);
      O[(((size_t)bb * NH + h) * SLEN + s) * DH + dd] = (_Float16)(accs[sub][r] + bvv);
    }
  }
}

// ---- score sub-tile, fully in registers ------------------------------------
// Returns per-lane v8f of final scores (Q.K^T + relpos bias)/8 + mask for the
// 8 C-layout elements (row 8g+r, col lane&15) of the 16x16 tile at (l0, r0).
// Bias diagonal gather done with ds_bpermute (no LDS memory, no barriers):
// element (m,n) needs dist row t0+m-n; it lives in lane ((m-n+15)&15)+16g of
// T1 (m<=n) or T2 (m>n), at the same C-layout VGPR index r.
__device__ __forceinline__ v8f score_tile(
    v16h qa0, v16h qa1, const _Float16* __restrict__ Kh,
    const _Float16* __restrict__ E16, const float* __restrict__ maskrow,
    int l0, int r0, int lane) {
  v8f c = {};
  c = wmma16(qa0, load_b_nk(Kh + (size_t)r0 * DH, DH, lane), c);
  c = wmma16(qa1, load_b_nk(Kh + (size_t)r0 * DH + 32, DH, lane), c);

  int t0 = l0 - r0 + (NPOS / 2);  // 2047
  v8f b1 = {}, b2 = {};
  b1 = wmma16(qa0, load_b_erows(E16, t0 - 15, 0, lane), b1);
  b1 = wmma16(qa1, load_b_erows(E16, t0 - 15, 32, lane), b1);
  b2 = wmma16(qa0, load_b_erows(E16, t0 + 1, 0, lane), b2);
  b2 = wmma16(qa1, load_b_erows(E16, t0 + 1, 32, lane), b2);

  int g = lane >> 4, n = lane & 15;
  float mv = maskrow[r0 + n];
  v8f sres;
#pragma unroll
  for (int r = 0; r < 8; ++r) {
    int m = 8 * g + r;
    int d = m - n;                           // in [-15, 15]
    int srcl = (((d + 15) & 15) + 16 * g) << 2;  // byte lane index for bpermute
    float p1 = __int_as_float(__builtin_amdgcn_ds_bpermute(srcl, __float_as_int(b1[r])));
    float p2 = __int_as_float(__builtin_amdgcn_ds_bpermute(srcl, __float_as_int(b2[r])));
    float bias = (d > 0) ? p2 : p1;
    sres[r] = (c[r] + bias) * 0.125f + mv;
  }
  return sres;
}

// ---- kernel 2: flash-style attention, one wave per 16-query tile -----------
// No __syncthreads anywhere: all LDS buffers are per-wave and per-wave LDS
// ops execute in order (DScnt), so the 4 waves per block run fully decoupled.
__global__ __launch_bounds__(128) void attn_kernel(
    const _Float16* __restrict__ Qw, const _Float16* __restrict__ Kw,
    const _Float16* __restrict__ Vw, const _Float16* __restrict__ V2w,
    const _Float16* __restrict__ E16, const float* __restrict__ mask,
    const float* __restrict__ posw, float* __restrict__ out) {
  __shared__ float    redA[4][256];
  __shared__ float    redB[4][256];
  __shared__ float    rowM[4][16];
  __shared__ float    rowIZ[4][16];
  __shared__ float    rowNZ[4][16];
  __shared__ _Float16 pts[4][512];
  __shared__ _Float16 nts[4][512];

  const int lane = threadIdx.x & 31, wv = threadIdx.x >> 5;
  const int g = lane >> 4, n = lane & 15;
  const int tile = blockIdx.x * 4 + wv;          // B*H*(S/16) tiles
  const int l0 = (tile & 127) * 16;
  const int bh = tile >> 7;
  const int b = bh >> 4, h = bh & 15;

  const size_t hoff = (size_t)bh * SLEN * DH;
  const _Float16* Qh = Qw + hoff;
  const _Float16* Kh = Kw + hoff;
  const _Float16* Vh = Vw + hoff;
  const _Float16* V2h = V2w + hoff;
  const float* maskrow = mask + (size_t)b * SLEN;

  float* rA = redA[wv];  float* rB = redB[wv];
  _Float16* ptw = pts[wv];  _Float16* ntw = nts[wv];

  v16h qa0 = load_a_frag(Qh + (size_t)l0 * DH, DH, lane);
  v16h qa1 = load_a_frag(Qh + (size_t)l0 * DH + 32, DH, lane);

  // ---- pass 1: per-lane online max/sum over this lane's column slice ------
  float ml[8], zl[8];
#pragma unroll
  for (int r = 0; r < 8; ++r) { ml[r] = -1e30f; zl[r] = 0.0f; }

  for (int r0 = 0; r0 < SLEN; r0 += 16) {
    v8f s = score_tile(qa0, qa1, Kh, E16, maskrow, l0, r0, lane);
#pragma unroll
    for (int r = 0; r < 8; ++r) {
      float mn = fmaxf(ml[r], s[r]);
      zl[r] = zl[r] * __expf(ml[r] - mn) + __expf(s[r] - mn);
      ml[r] = mn;
    }
  }
  // cross-lane merge: m_row = max_l m_l ; Z_row = sum_l z_l * exp(m_l - m_row)
#pragma unroll
  for (int r = 0; r < 8; ++r) {
    rA[(8 * g + r) * 16 + n] = ml[r];
    rB[(8 * g + r) * 16 + n] = zl[r];
  }
  if (lane < 16) {
    float mm = -1e30f;
#pragma unroll
    for (int j = 0; j < 16; ++j) mm = fmaxf(mm, rA[lane * 16 + j]);
    float Z = 0.0f;
#pragma unroll
    for (int j = 0; j < 16; ++j) Z += rB[lane * 16 + j] * __expf(rA[lane * 16 + j] - mm);
    rowM[wv][lane] = mm;
    rowIZ[wv][lane] = 1.0f / fmaxf(Z, 1e-30f);
  }
  float mreg[8], izreg[8];
#pragma unroll
  for (int r = 0; r < 8; ++r) {
    mreg[r] = rowM[wv][8 * g + r];
    izreg[r] = rowIZ[wv][8 * g + r];
  }

  // ---- pass 2: probs + neg weights in registers, WMMA accumulation --------
  const float NSC = 9.5367431640625e-07f;   // 2^-20: fit 1/(p+eps) in f16
  v8f pa0 = {}, pa1 = {}, pa2 = {}, pa3 = {};
  v8f na0 = {}, na1 = {}, na2 = {}, na3 = {};
  float nzp[8];
#pragma unroll
  for (int r = 0; r < 8; ++r) nzp[r] = 0.0f;

  for (int r0 = 0; r0 < SLEN; r0 += 32) {
#pragma unroll
    for (int hh = 0; hh < 2; ++hh) {
      int rc = r0 + hh * 16;
      v8f s = score_tile(qa0, qa1, Kh, E16, maskrow, l0, rc, lane);
      float mv = maskrow[rc + n];
      bool masked = (mv < -1.0f);
#pragma unroll
      for (int r = 0; r < 8; ++r) {
        int m = 8 * g + r;
        float p = __expf(s[r] - mreg[r]) * izreg[r];
        ptw[m * 32 + hh * 16 + n] = (_Float16)p;
        float nw = masked ? 0.0f : 1.0f / (p + 1e-9f);
        nzp[r] += nw;
        ntw[m * 32 + hh * 16 + n] = (_Float16)(nw * NSC);
      }
    }
    v16h pfa = load_a_frag(ptw, 32, lane);   // in-order per-wave LDS: safe
    v16h nfa = load_a_frag(ntw, 32, lane);
    pa0 = wmma16(pfa, load_b_kn(Vh  + (size_t)r0 * DH +  0, DH, lane), pa0);
    pa1 = wmma16(pfa, load_b_kn(Vh  + (size_t)r0 * DH + 16, DH, lane), pa1);
    pa2 = wmma16(pfa, load_b_kn(Vh  + (size_t)r0 * DH + 32, DH, lane), pa2);
    pa3 = wmma16(pfa, load_b_kn(Vh  + (size_t)r0 * DH + 48, DH, lane), pa3);
    na0 = wmma16(nfa, load_b_kn(V2h + (size_t)r0 * DH +  0, DH, lane), na0);
    na1 = wmma16(nfa, load_b_kn(V2h + (size_t)r0 * DH + 16, DH, lane), na1);
    na2 = wmma16(nfa, load_b_kn(V2h + (size_t)r0 * DH + 32, DH, lane), na2);
    na3 = wmma16(nfa, load_b_kn(V2h + (size_t)r0 * DH + 48, DH, lane), na3);
  }

  // ---- neg-weight row sums (cross-lane) -----------------------------------
#pragma unroll
  for (int r = 0; r < 8; ++r) rA[(8 * g + r) * 16 + n] = nzp[r];
  if (lane < 16) {
    float t = 0.0f;
#pragma unroll
    for (int j = 0; j < 16; ++j) t += rA[lane * 16 + j];
    rowNZ[wv][lane] = t;
  }

  float gg = 1.0f / (1.0f + __expf(-posw[h]));
  v8f pAcc[4] = {pa0, pa1, pa2, pa3};
  v8f nAcc[4] = {na0, na1, na2, na3};
#pragma unroll
  for (int sub = 0; sub < 4; ++sub) {
#pragma unroll
    for (int r = 0; r < 8; ++r) {
      int m = 8 * g + r;
      int s = l0 + m, dd = sub * 16 + n;
      float nz = fmaxf(rowNZ[wv][m], 1e-30f);
      float neg = nAcc[sub][r] * 1048576.0f / nz;  // undo 2^-20 scaling
      float val = gg * pAcc[sub][r] + (1.0f - gg) * neg;
      out[((size_t)b * SLEN + s) * HIDN + (size_t)h * DH + dd] = val;
    }
  }
}

extern "C" void kernel_launch(void* const* d_in, const int* in_sizes, int n_in,
                              void* d_out, int out_size, void* d_ws, size_t ws_size,
                              hipStream_t stream) {
  const float* X   = (const float*)d_in[0];
  const float* msk = (const float*)d_in[1];
  const float* Wq  = (const float*)d_in[2];
  const float* bq  = (const float*)d_in[3];
  const float* Wk  = (const float*)d_in[4];
  const float* bk  = (const float*)d_in[5];
  const float* Wv  = (const float*)d_in[6];
  const float* bv  = (const float*)d_in[7];
  const float* Wv2 = (const float*)d_in[8];
  const float* bv2 = (const float*)d_in[9];
  const float* dE  = (const float*)d_in[10];
  const float* pw  = (const float*)d_in[11];
  float* out = (float*)d_out;

  const size_t perBuf = (size_t)NB * NH * SLEN * DH;  // 4,194,304 halves each
  _Float16* Qw  = (_Float16*)d_ws;
  _Float16* Kw  = Qw + perBuf;
  _Float16* Vw  = Kw + perBuf;
  _Float16* V2w = Vw + perBuf;
  _Float16* E16 = V2w + perBuf;                       // NPOS*DH halves

  {
    int n = NPOS * DH;
    cvt_e_kernel<<<(n + 255) / 256, 256, 0, stream>>>(dE, E16, n);
  }
  {
    dim3 grd((NB * SLEN) / 64, HIDN / 64, 4);
    proj_kernel<<<grd, 128, 0, stream>>>(X, Wq, bq, Wk, bk, Wv, bv, Wv2, bv2,
                                         Qw, Kw, Vw, V2w);
  }
  {
    dim3 grd((NB * NH * (SLEN / 16)) / 4);
    attn_kernel<<<grd, 128, 0, stream>>>(Qw, Kw, Vw, V2w, E16, msk, pw, out);
  }
}
